// EncoderLayer_43946105372849
// MI455X (gfx1250) — matrix operand order repaired and checked by reading, not compiled, not run on hardware
//
#include <hip/hip_runtime.h>
#include <hip/hip_bf16.h>
#include <math.h>

typedef __attribute__((ext_vector_type(16))) __bf16 v16bf;
typedef __attribute__((ext_vector_type(8)))  __bf16 v8bf;
typedef __attribute__((ext_vector_type(4)))  __bf16 v4bf;
typedef __attribute__((ext_vector_type(8)))  float  v8f;
typedef __attribute__((ext_vector_type(4)))  float  f32x4;

union FragU { v16bf v; struct { v8bf lo, hi; } p; };

static constexpr int  Bsz = 2, S = 2048, D = 256, F = 1024, NL = 10, H = 8, DH = 32;
static constexpr int  MROWS = Bsz * S;            // 4096
static constexpr long NSD   = (long)Bsz * S * D;  // 1048576

// ---------------------------------------------------------------------------
// Generic bf16-WMMA GEMM:  C = act(alpha * A(MxK) * B + bias)
// Template: BTRANS = B stored [N,K] (1) vs [K,N] (0); NG = N-guard (only the
// AV GEMM, N=32 < 64, needs it). Requirements met by all call sites:
//   M % 128 == 0, K % 32 == 0, N % 4 == 0, 16B-aligned pointers.
// Block tile 128x64, 8 waves, wave tile 32x32 (2x2 WMMA 16x16x32 tiles).
// Double-buffered LDS: one barrier per k-step; global loads for step t+1 are
// issued before the WMMAs of step t.
// ---------------------------------------------------------------------------
template <int BTRANS, int NG>
__global__ __launch_bounds__(256)
void gemm_bf16_wmma(const float* __restrict__ A, const float* __restrict__ Bm,
                    const float* __restrict__ bias, float* __restrict__ C,
                    int M, int N, int K, int lda, int ldb, int ldc,
                    int Hdim, long sAb, long sAh, long sBb, long sBh,
                    long sCb, long sCh, float alpha, int relu)
{
    const int tid  = threadIdx.x;
    const int lane = tid & 31;
    const int wid  = tid >> 5;
    const int bz = blockIdx.z / Hdim;
    const int hz = blockIdx.z % Hdim;
    const long offA = (long)bz * sAb + (long)hz * sAh;
    const long offB = (long)bz * sBb + (long)hz * sBh;
    const long offC = (long)bz * sCb + (long)hz * sCh;
    const int m0 = blockIdx.y * 128;
    const int n0 = blockIdx.x * 64;
    const int wm = (wid >> 1) * 32;   // wave row offset within block tile
    const int wn = (wid & 1) * 32;    // wave col offset within block tile

    __shared__ __align__(16) __bf16 lA[2][128][32];  // 16 KB
    __shared__ __align__(16) __bf16 lB[2][64][32];   //  8 KB

    // per-thread staging coordinates (quad = float4)
    int ar[4], ac4[4];                 // A: 4 quads (128 rows x 8 quad-cols)
    #pragma unroll
    for (int i = 0; i < 4; ++i) { int qi = tid + i * 256; ar[i] = qi >> 3; ac4[i] = (qi & 7) * 4; }
    int bx[2], by4[2];                 // B: 2 quads
    #pragma unroll
    for (int i = 0; i < 2; ++i) {
        int qi = tid + i * 256;
        if (BTRANS) { bx[i] = qi >> 3; by4[i] = (qi & 7) * 4; }   // [n row][k quad]
        else        { bx[i] = qi >> 4; by4[i] = (qi & 15) * 4; }  // [k row][n quad]
    }

    f32x4 aq[4], bq[2];

    auto load_tiles = [&](int k0) {
        #pragma unroll
        for (int i = 0; i < 4; ++i)
            aq[i] = *(const f32x4*)&A[offA + (long)(m0 + ar[i]) * lda + (k0 + ac4[i])];
        #pragma unroll
        for (int i = 0; i < 2; ++i) {
            if (BTRANS) {
                bq[i] = *(const f32x4*)&Bm[offB + (long)(n0 + bx[i]) * ldb + (k0 + by4[i])];
            } else {
                int gn = n0 + by4[i];
                if (NG) gn = (gn + 4 <= N) ? gn : (N - 4);        // clamped, in range
                bq[i] = *(const f32x4*)&Bm[offB + (long)(k0 + bx[i]) * ldb + gn];
            }
        }
    };

    auto store_tiles = [&](int buf) {
        #pragma unroll
        for (int i = 0; i < 4; ++i) {
            v4bf t;
            #pragma unroll
            for (int j = 0; j < 4; ++j) t[j] = (__bf16)aq[i][j];
            *(v4bf*)&lA[buf][ar[i]][ac4[i]] = t;                  // ds_store_b64
        }
        #pragma unroll
        for (int i = 0; i < 2; ++i) {
            if (BTRANS) {
                v4bf t;
                #pragma unroll
                for (int j = 0; j < 4; ++j) t[j] = (__bf16)bq[i][j];
                *(v4bf*)&lB[buf][bx[i]][by4[i]] = t;
            } else {
                bool ok = !NG || (n0 + by4[i] + 4) <= N;          // quad all-valid or all-OOB
                #pragma unroll
                for (int j = 0; j < 4; ++j)
                    lB[buf][by4[i] + j][bx[i]] = (__bf16)(ok ? bq[i][j] : 0.0f);
            }
        }
    };

    v8f acc[2][2];
    #pragma unroll
    for (int i = 0; i < 2; ++i)
        #pragma unroll
        for (int j = 0; j < 2; ++j)
            #pragma unroll
            for (int r = 0; r < 8; ++r) acc[i][j][r] = 0.0f;

    const int ml  = lane & 15;
    const int hiL = lane >> 4;

    load_tiles(0);
    store_tiles(0);
    __syncthreads();

    for (int k0 = 0; k0 < K; k0 += 32) {
        const int  cur  = (k0 >> 5) & 1;
        const bool more = (k0 + 32) < K;
        if (more) load_tiles(k0 + 32);            // global loads in flight

        v16bf afr[2], bfr[2];
        #pragma unroll
        for (int i = 0; i < 2; ++i) {
            FragU ua;
            const __bf16* pa = &lA[cur][wm + i * 16 + ml][hiL * 8]; // k=8h..+7, 16+8h..+7
            ua.p.lo = *(const v8bf*)pa;
            ua.p.hi = *(const v8bf*)(pa + 16);
            afr[i] = ua.v;
            FragU ub;
            const __bf16* pb = &lB[cur][wn + i * 16 + ml][hiL * 8]; // k=8h..+7, 16+8h..+7
            ub.p.lo = *(const v8bf*)pb;
            ub.p.hi = *(const v8bf*)(pb + 16);
            bfr[i] = ub.v;
        }
        if (more) store_tiles(cur ^ 1);           // write other buffer, no barrier needed

        #pragma unroll
        for (int i = 0; i < 2; ++i)
            #pragma unroll
            for (int j = 0; j < 2; ++j)
                acc[i][j] = __builtin_amdgcn_wmma_f32_16x16x32_bf16(
                    false, afr[i], false, bfr[j], (short)0, acc[i][j], false, false);
        __syncthreads();                          // single barrier per k-step
    }

    // ---- epilogue: alpha, bias, relu, store ----
    #pragma unroll
    for (int i = 0; i < 2; ++i) {
        #pragma unroll
        for (int j = 0; j < 2; ++j) {
            int n = n0 + wn + j * 16 + (lane & 15);
            if (NG && n >= N) continue;
            float bv = bias ? bias[n] : 0.0f;
            #pragma unroll
            for (int r = 0; r < 8; ++r) {
                int m = m0 + wm + i * 16 + r + ((lane >> 4) << 3);
                float vv = acc[i][j][r] * alpha + bv;
                if (relu) vv = fmaxf(vv, 0.0f);
                C[offC + (long)m * ldc + n] = vv;
            }
        }
    }
}

// ---------------------------------------------------------------------------
// Build wavelet difference-filters on device (they are not kernel inputs).
// filt[si*64 + m], m in [0, L], L = 10*s+1; filt = -sqrt(s)*(k[m-1]-k[m]).
// ---------------------------------------------------------------------------
__global__ void build_filters(float* __restrict__ filt)
{
    __shared__ double ip[1024];
    const int tid = threadIdx.x;
    const double step = 10.0 / 1023.0;
    const double amp  = 0.8932438417380023;   // (2/pi)^0.25
    for (int i = tid; i < 1024; i += 256) {
        double xi = -5.0 + step * (double)i;
        ip[i] = -2.0 * xi * amp * exp(-xi * xi);
    }
    __syncthreads();
    if (tid == 0) {
        double a = 0.0;
        for (int i = 0; i < 1024; ++i) { a += ip[i]; ip[i] = a * step; }
    }
    __syncthreads();
    const int scales[3] = {1, 3, 6};
    for (int si = 0; si < 3; ++si) {
        int s = scales[si];
        int L = 10 * s + 1;
        for (int m = tid; m <= L; m += 256) {
            float km1 = 0.0f, km = 0.0f;
            if (m > 0) { int j = (int)((double)(m - 1) / ((double)s * step)); if (j > 1023) j = 1023; km1 = (float)ip[j]; }
            if (m < L) { int j = (int)((double)m       / ((double)s * step)); if (j > 1023) j = 1023; km  = (float)ip[j]; }
            filt[si * 64 + m] = -sqrtf((float)s) * (km1 - km);
        }
    }
}

// out[b,n,c] = sum_m filt[m] * x[b, n+off+m, c]   (zero-padded)
__global__ __launch_bounds__(256)
void cwt_apply(const float* __restrict__ x, const float* __restrict__ filt,
               float* __restrict__ out, int taps, int off)
{
    long idx = (long)blockIdx.x * 256 + threadIdx.x;
    int  c = (int)(idx & (D - 1));
    int  n = (int)((idx >> 8) & (S - 1));
    long b = idx >> 19;
    float acc = 0.0f;
    for (int m = 0; m < taps; ++m) {
        int sr = n + off + m;
        if (sr >= 0 && sr < S)
            acc += filt[m] * x[((b << 11) + sr) * D + c];
    }
    out[idx] = acc;
}

// row softmax over 2048 cols, in place (rows = B*H*S)
__global__ __launch_bounds__(256)
void softmax2048(float* __restrict__ data)
{
    long base = (long)blockIdx.x * S;
    int tid = threadIdx.x;
    float r[8];
    float mx = -3.4e38f;
    #pragma unroll
    for (int i = 0; i < 8; ++i) { r[i] = data[base + tid + i * 256]; mx = fmaxf(mx, r[i]); }
    __shared__ float red[256];
    red[tid] = mx; __syncthreads();
    for (int s = 128; s > 0; s >>= 1) { if (tid < s) red[tid] = fmaxf(red[tid], red[tid + s]); __syncthreads(); }
    mx = red[0]; __syncthreads();
    float sum = 0.0f;
    #pragma unroll
    for (int i = 0; i < 8; ++i) { r[i] = __expf(r[i] - mx); sum += r[i]; }
    red[tid] = sum; __syncthreads();
    for (int s = 128; s > 0; s >>= 1) { if (tid < s) red[tid] += red[tid + s]; __syncthreads(); }
    float inv = 1.0f / red[0];
    #pragma unroll
    for (int i = 0; i < 8; ++i) data[base + tid + i * 256] = r[i] * inv;
}

// xm (+)= LayerNorm(2*sx + ao)/3   (one row of D=256 per block)
__global__ __launch_bounds__(256)
void ln1_accum(const float* __restrict__ sx, const float* __restrict__ ao,
               const float* __restrict__ g, const float* __restrict__ be,
               float* __restrict__ xm, int init)
{
    long base = (long)blockIdx.x * D;
    int c = threadIdx.x;
    float v = 2.0f * sx[base + c] + ao[base + c];
    __shared__ float s1[256], s2[256];
    s1[c] = v; s2[c] = v * v; __syncthreads();
    for (int s = 128; s > 0; s >>= 1) { if (c < s) { s1[c] += s1[c + s]; s2[c] += s2[c + s]; } __syncthreads(); }
    float mean = s1[0] * (1.0f / 256.0f);
    float var  = s2[0] * (1.0f / 256.0f) - mean * mean;
    float o = (v - mean) * rsqrtf(var + 1e-5f) * g[c] + be[c];
    o *= (1.0f / 3.0f);
    xm[base + c] = init ? o : (xm[base + c] + o);
}

// out = LayerNorm(xm + y)
__global__ __launch_bounds__(256)
void ln_final(const float* __restrict__ xm, const float* __restrict__ y,
              const float* __restrict__ g, const float* __restrict__ be,
              float* __restrict__ out)
{
    long base = (long)blockIdx.x * D;
    int c = threadIdx.x;
    float v = xm[base + c] + y[base + c];
    __shared__ float s1[256], s2[256];
    s1[c] = v; s2[c] = v * v; __syncthreads();
    for (int s = 128; s > 0; s >>= 1) { if (c < s) { s1[c] += s1[c + s]; s2[c] += s2[c + s]; } __syncthreads(); }
    float mean = s1[0] * (1.0f / 256.0f);
    float var  = s2[0] * (1.0f / 256.0f) - mean * mean;
    out[base + c] = (v - mean) * rsqrtf(var + 1e-5f) * g[c] + be[c];
}

// xcol[(b,s), d*3+t] = xm[b, s+2t-2, d]  (dilated-conv im2col, K = 768)
__global__ __launch_bounds__(256)
void im2col3(const float* __restrict__ xm, float* __restrict__ xcol)
{
    long idx = (long)blockIdx.x * 256 + threadIdx.x;
    int  col = (int)(idx % 768);
    long row = idx / 768;
    int  d = col / 3, t = col % 3;
    long b = row >> 11;
    int  s = (int)(row & (S - 1));
    int  ss = s + 2 * t - 2;
    float v = 0.0f;
    if (ss >= 0 && ss < S) v = xm[((b << 11) + ss) * D + d];
    xcol[idx] = v;
}

// ---------------------------------------------------------------------------
extern "C" void kernel_launch(void* const* d_in, const int* in_sizes, int n_in,
                              void* d_out, int out_size, void* d_ws, size_t ws_size,
                              hipStream_t stream)
{
    const float* x   = (const float*)d_in[0];
    const float* Wt  = (const float*)d_in[1];
    const float* bt  = (const float*)d_in[2];
    const float* Wq  = (const float*)d_in[3];
    const float* bq  = (const float*)d_in[4];
    const float* Wk  = (const float*)d_in[5];
    const float* bk  = (const float*)d_in[6];
    const float* Wv  = (const float*)d_in[7];
    const float* bv  = (const float*)d_in[8];
    const float* Wo  = (const float*)d_in[9];
    const float* bo  = (const float*)d_in[10];
    const float* g1  = (const float*)d_in[11];
    const float* be1 = (const float*)d_in[12];
    const float* W1  = (const float*)d_in[13];   // (F, D, 3) -> flat [F][768]
    const float* b1  = (const float*)d_in[14];
    const float* W2  = (const float*)d_in[15];   // (D, F)
    const float* b2  = (const float*)d_in[16];
    const float* g2  = (const float*)d_in[17];
    const float* be2 = (const float*)d_in[18];
    // d_in[19] = n_heads (known: 8)

    float* out  = (float*)d_out;
    float* attn = out + NSD;                     // (B,H,S,S) region of d_out

    // workspace layout (~48 MiB of f32)
    float* w    = (float*)d_ws;
    float* filt = w;                             // 3*64
    float* sxA  = w + 256;
    float* sxB  = sxA + NSD;
    float* qb   = sxB + NSD;
    float* kb   = qb  + NSD;
    float* vb   = kb  + NSD;
    float* aoh  = vb  + NSD;
    float* aob  = aoh + NSD;
    float* xm   = aob + NSD;
    float* hbuf = xm  + NSD;                     // 4096*1024
    float* xcol = qb;                            // reuse q|k|v = 4096*768
    float* ybuf = sxB;                           // reuse

    // btrans=1: B stored [N,K]; guard=1 only for the N=32 AV GEMM
    auto gemm = [&](int btrans, const float* A, const float* Bm, const float* bias, float* C,
                    int M, int N, int K, int lda, int ldb, int ldc,
                    int relu, float alpha = 1.0f, int batch = 1, int Hdim = 1,
                    long sAb = 0, long sAh = 0, long sBb = 0, long sBh = 0,
                    long sCb = 0, long sCh = 0) {
        dim3 grid((N + 63) / 64, (M + 127) / 128, batch);
        if (btrans)
            gemm_bf16_wmma<1, 0><<<grid, 256, 0, stream>>>(A, Bm, bias, C, M, N, K, lda, ldb, ldc,
                Hdim, sAb, sAh, sBb, sBh, sCb, sCh, alpha, relu);
        else if ((N & 63) == 0)
            gemm_bf16_wmma<0, 0><<<grid, 256, 0, stream>>>(A, Bm, bias, C, M, N, K, lda, ldb, ldc,
                Hdim, sAb, sAh, sBb, sBh, sCb, sCh, alpha, relu);
        else
            gemm_bf16_wmma<0, 1><<<grid, 256, 0, stream>>>(A, Bm, bias, C, M, N, K, lda, ldb, ldc,
                Hdim, sAb, sAh, sBb, sBh, sCb, sCh, alpha, relu);
    };

    build_filters<<<1, 256, 0, stream>>>(filt);

    const int taps[3] = {12, 32, 62};    // L+1
    const int offs[3] = {-6, -16, -31};  // lo - L + 1

    for (int si = 0; si < 3; ++si) {
        cwt_apply<<<(int)(NSD / 256), 256, 0, stream>>>(x, filt + si * 64, sxA, taps[si], offs[si]);

        // 10-layer ReLU dense stack (ping-pong; even count -> ends in sxA)
        float* cur = sxA; float* nxt = sxB;
        for (int l = 0; l < NL; ++l) {
            gemm(1, cur, Wt + (long)l * D * D, bt + (long)l * D, nxt,
                 MROWS, D, D, D, D, D, /*relu*/1);
            float* t = cur; cur = nxt; nxt = t;
        }

        gemm(0, cur, Wq, bq, qb, MROWS, D, D, D, D, D, 0);
        gemm(0, cur, Wk, bk, kb, MROWS, D, D, D, D, D, 0);
        gemm(0, cur, Wv, bv, vb, MROWS, D, D, D, D, D, 0);

        // scores = alpha * Q K^T  -> written straight into d_out attn region
        gemm(1, qb, kb, nullptr, attn, S, S, DH, D, D, S, 0,
             0.17677669529663689f, Bsz * H, H,
             (long)S * D, DH, (long)S * D, DH, (long)H * S * S, (long)S * S);
        softmax2048<<<Bsz * H * S, 256, 0, stream>>>(attn);

        // AV: (S x S) @ (S x 32) per (b,h)  (N=32 -> guarded variant)
        gemm(0, attn, vb, nullptr, aoh, S, DH, S, S, D, D, 0, 1.0f, Bsz * H, H,
             (long)H * S * S, (long)S * S, (long)S * D, DH, (long)S * D, DH);

        gemm(0, aoh, Wo, bo, aob, MROWS, D, D, D, D, D, 0);
        ln1_accum<<<MROWS, 256, 0, stream>>>(cur, aob, g1, be1, xm, si == 0 ? 1 : 0);
    }

    // FFN: dilated conv as im2col + GEMM, then W2, then final LN into d_out
    im2col3<<<(MROWS * 768) / 256, 256, 0, stream>>>(xm, xcol);
    gemm(1, xcol, W1, b1, hbuf, MROWS, F, 3 * D, 3 * D, 3 * D, F, /*relu*/1);
    gemm(1, hbuf, W2, b2, ybuf, MROWS, D, F, F, F, D, /*relu*/1);
    ln_final<<<MROWS, 256, 0, stream>>>(xm, ybuf, g2, be2, out);

    (void)in_sizes; (void)n_in; (void)out_size; (void)ws_size;
}
